// OscSSMCell_11166914969867
// MI455X (gfx1250) — matrix operand order repaired and checked by reading, not compiled, not run on hardware
//
#include <hip/hip_runtime.h>

typedef __attribute__((ext_vector_type(16))) __bf16 v16bf;
typedef __attribute__((ext_vector_type(8)))  float  v8f;

#define IN_DIM   512
#define HID_DIM  1024
#define MWG      32
#define UA_STR   520   /* uA row stride in bf16 elems: 1040B -> 260 dwords, 260%64=4 -> conflict-free b128 reads */
#define HA_STR   1032  /* hA row stride: 2064B -> 516 dwords, 516%64=4 */

__device__ inline v8f wmma_bf16(v16bf a, v16bf b, v8f c) {
  // D = A(16x32 bf16) * B(32x16 bf16) + C(16x16 f32)
  return __builtin_amdgcn_wmma_f32_16x16x32_bf16(false, a, false, b, (short)0, c, false, false);
}

// A fragment (16x32, MxK) from LDS, row-major bf16.
// Lane l: row = m0 + (l&15); elements e -> K = kb+e, kb = (l>>4)*16.
__device__ inline v16bf load_a_lds(const __bf16* base, int rowstride, int m0, int k0, int lane) {
  const int row = m0 + (lane & 15);
  const int kb  = (lane >> 4) * 16;
  const __bf16* p = base + row * rowstride + k0 + kb;
  v16bf a;
#pragma unroll
  for (int e = 0; e < 16; ++e) { a[e] = p[e]; }
  return a;
}

// B fragment (32x16, KxN) from global bf16 weights stored [col][k] row-major (k contiguous).
// Lane l: col = n0 + (l&15); elements e -> K = (l>>4)*16 + e  (32 contiguous bytes per lane).
__device__ inline v16bf load_b_gl(const __bf16* __restrict__ W, int kstride, int n0, int k0, int lane) {
  const int col = n0 + (lane & 15);
  const __bf16* p = W + (size_t)col * kstride + k0 + ((lane >> 4) * 16);
  v16bf b;
#pragma unroll
  for (int e = 0; e < 16; ++e) b[e] = p[e];
  return b;
}

__device__ inline float fast_silu(float x) { return x / (1.0f + __expf(-x)); }

__global__ void __launch_bounds__(1024) prep_kernel(
    const float* __restrict__ W1, const float* __restrict__ W2,
    const float* __restrict__ Bmat, const float* __restrict__ log_alpha,
    const float* __restrict__ omega,
    __bf16* __restrict__ W1b, __bf16* __restrict__ W2b, __bf16* __restrict__ Bb,
    float* __restrict__ dcos, float* __restrict__ dsin) {
  const int i = blockIdx.x * blockDim.x + threadIdx.x;
  if (i < HID_DIM * HID_DIM) W2b[i] = (__bf16)W2[i];
  if (i < HID_DIM * IN_DIM) {
    W1b[i] = (__bf16)W1[i];                       // W1 already [j][k]
    const int j = i >> 9, k = i & 511;            // Bb[j][k] = Bmat[k][j] (transpose)
    Bb[i] = (__bf16)Bmat[(size_t)k * HID_DIM + j];
  }
  if (i < HID_DIM / 2) {
    const float la = log_alpha[i];
    const float alpha = (la > 20.0f) ? la : log1pf(__expf(la));  // softplus
    const float d = __expf(-alpha);
    dcos[i] = d * cosf(omega[i]);
    dsin[i] = d * sinf(omega[i]);
  }
}

__global__ void __launch_bounds__(1024) osc_fused_kernel(
    const float* __restrict__ u, const float* __restrict__ x,
    const __bf16* __restrict__ W1b, const __bf16* __restrict__ W2b,
    const __bf16* __restrict__ Bb,
    const float* __restrict__ b1, const float* __restrict__ b2,
    const float* __restrict__ bias, const float* __restrict__ scale,
    const float* __restrict__ dcos, const float* __restrict__ dsin,
    float* __restrict__ out) {
  extern __shared__ char smem[];
  __bf16* uA = (__bf16*)smem;                                   // [32][UA_STR]
  __bf16* hA = (__bf16*)(smem + MWG * UA_STR * 2);              // [32][HA_STR]
  float* rowsum = (float*)(smem + MWG * UA_STR * 2 + MWG * HA_STR * 2); // [32]

  const int tid  = threadIdx.x;
  const int lane = tid & 31;
  const int wave = tid >> 5;
  const int r0   = blockIdx.x * MWG;

  if (tid < MWG) rowsum[tid] = 0.0f;

  // ---- Stage 1: u tile (32x512 f32) -> bf16 LDS ----
  {
    const float4* uv = (const float4*)(u + (size_t)r0 * IN_DIM);
#pragma unroll
    for (int i = tid; i < MWG * IN_DIM / 4; i += 1024) {
      const float4 v = uv[i];
      const int row = i >> 7;              // 128 float4 per row
      const int cb  = (i & 127) << 2;
      __bf16* d = uA + row * UA_STR + cb;
      d[0] = (__bf16)v.x; d[1] = (__bf16)v.y; d[2] = (__bf16)v.z; d[3] = (__bf16)v.w;
    }
  }
  __syncthreads();

  // wave tiling: 32 waves = 2 M-groups x 16 N-groups; each wave: 1 M-tile x 4 N-tiles
  const int mg = wave >> 4;          // 0..1
  const int ng = wave & 15;          // 0..15
  const int mbase = mg * 16;         // local row base (one 16-row tile)
  const int nbase = ng * 64;         // col base (4 tiles of 16)

  const v8f vzero = {0, 0, 0, 0, 0, 0, 0, 0};
  v8f acc1[4], acc3[4];              // tile index = nt
#pragma unroll
  for (int t = 0; t < 4; ++t) { acc1[t] = vzero; acc3[t] = vzero; }

  // ---- Stage 2: fused GEMM1 (h-pre = u@W1^T) and GEMM3 (Bu = u@Bmat), K = 512 ----
  for (int k0 = 0; k0 < IN_DIM; k0 += 32) {
    const v16bf a0 = load_a_lds(uA, UA_STR, mbase, k0, lane);
#pragma unroll
    for (int nt = 0; nt < 4; ++nt) {
      const v16bf bw = load_b_gl(W1b, IN_DIM, nbase + nt * 16, k0, lane);
      acc1[nt] = wmma_bf16(a0, bw, acc1[nt]);
      const v16bf bb = load_b_gl(Bb, IN_DIM, nbase + nt * 16, k0, lane);
      acc3[nt] = wmma_bf16(a0, bb, acc3[nt]);
    }
  }

  // h = silu(acc1 + b1) -> bf16 LDS (C/D layout: row = (lane>>4)*8 + r, col = lane&15)
  const int hi = (lane >> 4) * 8;
  const int cl = lane & 15;
#pragma unroll
  for (int nt = 0; nt < 4; ++nt) {
    const int col = nbase + nt * 16 + cl;
    const float b1c = b1[col];
#pragma unroll
    for (int r = 0; r < 8; ++r) {
      const int lrow = mbase + hi + r;
      const float h = fast_silu(acc1[nt][r] + b1c);
      hA[lrow * HA_STR + col] = (__bf16)h;
    }
  }
  __syncthreads();

  // ---- Stage 3: GEMM2 (mod-pre = h@W2^T), K = 1024 ----
  v8f acc2[4];
#pragma unroll
  for (int t = 0; t < 4; ++t) acc2[t] = vzero;
  for (int k0 = 0; k0 < HID_DIM; k0 += 32) {
    const v16bf a0 = load_a_lds(hA, HA_STR, mbase, k0, lane);
#pragma unroll
    for (int nt = 0; nt < 4; ++nt) {
      const v16bf bw = load_b_gl(W2b, HID_DIM, nbase + nt * 16, k0, lane);
      acc2[nt] = wmma_bf16(a0, bw, acc2[nt]);
    }
  }

  // ---- Stage 4: epilogue (rotation, gate, silu, row sum of s^2) ----
  float ps[8];
#pragma unroll
  for (int i = 0; i < 8; ++i) ps[i] = 0.0f;

#pragma unroll
  for (int nt = 0; nt < 4; ++nt) {
    const int col  = nbase + nt * 16 + cl;
    const float b2c  = b2[col];
    const float bic  = bias[col];
    const int   m    = col >> 1;
    const float dc   = dcos[m];
    const float dsn  = dsin[m];
    const bool  even = (col & 1) == 0;
#pragma unroll
    for (int r = 0; r < 8; ++r) {
      const int lrow = mbase + hi + r;
      const size_t grow = (size_t)(r0 + lrow);
      const float2 xv = *(const float2*)(x + grow * HID_DIM + (col & ~1));
      const float xr = even ? (dc * xv.x - dsn * xv.y) : (dsn * xv.x + dc * xv.y);
      const float mod = tanhf(acc2[nt][r] + b2c);
      const float xn = xr + (1.0f + mod) * acc3[nt][r] + bic;
      const float s = fast_silu(xn);
      acc3[nt][r] = s;                 // keep s for final write
      ps[r] += s * s;
    }
  }

  // reduce s^2 partials across the 16 lanes that share each row
#pragma unroll
  for (int o = 1; o < 16; o <<= 1) {
#pragma unroll
    for (int i = 0; i < 8; ++i) ps[i] += __shfl_xor(ps[i], o, 32);
  }
  if (cl == 0) {
#pragma unroll
    for (int r = 0; r < 8; ++r)
      atomicAdd(&rowsum[mbase + hi + r], ps[r]);
  }
  __syncthreads();

  // ---- Stage 5: RMSNorm + store ----
#pragma unroll
  for (int nt = 0; nt < 4; ++nt) {
    const int col = nbase + nt * 16 + cl;
    const float sc = scale[col];
#pragma unroll
    for (int r = 0; r < 8; ++r) {
      const int lrow = mbase + hi + r;
      const size_t grow = (size_t)(r0 + lrow);
      const float inv = rsqrtf(rowsum[lrow] * (1.0f / (float)HID_DIM) + 1e-5f);
      out[grow * HID_DIM + col] = sc * acc3[nt][r] * inv;
    }
  }
}

extern "C" void kernel_launch(void* const* d_in, const int* in_sizes, int n_in,
                              void* d_out, int out_size, void* d_ws, size_t ws_size,
                              hipStream_t stream) {
  const float* u_t       = (const float*)d_in[0];
  const float* x_t       = (const float*)d_in[1];
  const float* log_alpha = (const float*)d_in[2];
  const float* omega     = (const float*)d_in[3];
  const float* Bmat      = (const float*)d_in[4];
  const float* bias      = (const float*)d_in[5];
  const float* W1        = (const float*)d_in[6];
  const float* b1        = (const float*)d_in[7];
  const float* W2        = (const float*)d_in[8];
  const float* b2        = (const float*)d_in[9];
  const float* scale     = (const float*)d_in[10];
  float* out = (float*)d_out;

  const int Bsz = in_sizes[0] / IN_DIM;

  // workspace layout (bf16 weights + rotation constants)
  char* ws = (char*)d_ws;
  __bf16* W1b = (__bf16*)ws;                                           // 1 MB
  __bf16* W2b = (__bf16*)(ws + (size_t)HID_DIM * IN_DIM * 2);          // 2 MB
  __bf16* Bb  = (__bf16*)(ws + (size_t)HID_DIM * IN_DIM * 2
                             + (size_t)HID_DIM * HID_DIM * 2);         // 1 MB
  float* dcos = (float*)(ws + (size_t)HID_DIM * IN_DIM * 4
                            + (size_t)HID_DIM * HID_DIM * 2);
  float* dsin = dcos + HID_DIM / 2;

  const int prep_n = HID_DIM * HID_DIM;
  prep_kernel<<<(prep_n + 1023) / 1024, 1024, 0, stream>>>(
      W1, W2, Bmat, log_alpha, omega, W1b, W2b, Bb, dcos, dsin);

  const size_t lds_bytes = (size_t)(MWG * UA_STR + MWG * HA_STR) * 2 + MWG * 4;
  osc_fused_kernel<<<Bsz / MWG, 1024, lds_bytes, stream>>>(
      u_t, x_t, W1b, W2b, Bb, b1, b2, bias, scale, dcos, dsin, out);
}
